// CausalSelfAttention_10737418240409
// MI455X (gfx1250) — matrix operand order repaired and checked by reading, not compiled, not run on hardware
//
#include <hip/hip_runtime.h>

// ---------------------------------------------------------------------------
// Types
// ---------------------------------------------------------------------------
typedef __attribute__((ext_vector_type(16))) __bf16         bf16x16;
typedef __attribute__((ext_vector_type(8)))  __bf16         bf16x8;
typedef __attribute__((ext_vector_type(8)))  float          f32x8;
typedef __attribute__((ext_vector_type(8)))  unsigned short u16x8;

#define BATCH 4
#define SEQ   2048
#define EMB   1024
#define NHEAD 16
#define DHEAD 64
#define MROWS (BATCH * SEQ)   // 8192

// ---------------------------------------------------------------------------
// Helpers
// ---------------------------------------------------------------------------
static __device__ inline unsigned short f2bf_bits(float f) {
  unsigned u = __float_as_uint(f);
  u += 0x7FFFu + ((u >> 16) & 1u);          // round-to-nearest-even
  return (unsigned short)(u >> 16);
}
static __device__ inline __bf16 f2bf(float f) {
  unsigned short h = f2bf_bits(f);
  return __builtin_bit_cast(__bf16, h);
}
static __device__ inline bf16x16 cat8(bf16x8 lo, bf16x8 hi) {
  bf16x16 r;
#pragma unroll
  for (int i = 0; i < 8; ++i) { r[i] = lo[i]; r[i + 8] = hi[i]; }
  return r;
}
static __device__ inline f32x8 wmma_bf16(bf16x16 a, bf16x16 b, f32x8 c) {
  return __builtin_amdgcn_wmma_f32_16x16x32_bf16(
      false, a, false, b, (short)0, c, false, false);
}

// ---------------------------------------------------------------------------
// fp32 -> bf16 conversion, 8 elements per thread (all sizes divisible by 8)
// ---------------------------------------------------------------------------
__global__ void cvt_f32_bf16_v8(const float* __restrict__ in,
                                unsigned short* __restrict__ out, int n8) {
  int i = blockIdx.x * blockDim.x + threadIdx.x;
  if (i < n8) {
    f32x8 v = *(const f32x8*)(in + (size_t)i * 8);
    u16x8 o;
#pragma unroll
    for (int j = 0; j < 8; ++j) o[j] = f2bf_bits(v[j]);
    *(u16x8*)(out + (size_t)i * 8) = o;
  }
}

// ---------------------------------------------------------------------------
// GEMM: C[M,N] = oscale * (A[M,K] * W[N,K]^T)   (bf16 in, f32 accumulate)
// 256 threads = 8 waves, block tile 128x256, each wave a 64x64 tile.
// All 8 fragment loads of a K-step are issued before the 16 WMMAs so the
// scheduler can drain them with partial s_wait_loadcnt.
// mode 0: bf16 row-major [M,1024]
// mode 1: bf16 per-head transposed [BATCH,NHEAD,DHEAD,SEQ]  (for V)
// mode 2: f32  row-major [M,1024]  (final projection into d_out)
// ---------------------------------------------------------------------------
__global__ __launch_bounds__(256)
void gemm_bf16(const __bf16* __restrict__ A, const __bf16* __restrict__ Bw,
               void* __restrict__ Out, int mode, float oscale) {
  const int lane  = threadIdx.x & 31;
  const int wave  = threadIdx.x >> 5;
  const int lr    = lane & 15;
  const int lh    = lane >> 4;
  const int mtile = blockIdx.y * 128 + (wave >> 2) * 64;
  const int ntile = blockIdx.x * 256 + (wave & 3) * 64;

  f32x8 acc[4][4];
#pragma unroll
  for (int mi = 0; mi < 4; ++mi)
#pragma unroll
    for (int ni = 0; ni < 4; ++ni)
#pragma unroll
      for (int r = 0; r < 8; ++r) acc[mi][ni][r] = 0.f;

  // A fragment (16x32 bf16): lane holds row (mtile+mi*16+lr),
//   chunks at K = k + lh*16 and K = k + lh*16 + 8   (two 16B loads)
  // B fragment (32x16): lane holds col (ntile+ni*16+lr),
  //   chunk at K = k + lh*16 (one 32B load) from W stored [N,K]
  const __bf16* aB = A  + (size_t)(mtile + lr) * EMB + lh * 16;
  const __bf16* bB = Bw + (size_t)(ntile + lr) * EMB + lh * 16;

  for (int k = 0; k < EMB; k += 32) {
    if (k + 128 < EMB) {                       // gfx1250 global_prefetch
      __builtin_prefetch(aB + k + 128, 0, 3);
      __builtin_prefetch(bB + k + 128, 0, 3);
    }
    bf16x16 af[4], bfq[4];
#pragma unroll
    for (int mi = 0; mi < 4; ++mi) {
      const __bf16* p = aB + (size_t)mi * 16 * EMB + k;
      af[mi] = cat8(*(const bf16x8*)p, *(const bf16x8*)(p + 8));
    }
#pragma unroll
    for (int ni = 0; ni < 4; ++ni)
      bfq[ni] = *(const bf16x16*)(bB + (size_t)ni * 16 * EMB + k);
#pragma unroll
    for (int ni = 0; ni < 4; ++ni)
#pragma unroll
      for (int mi = 0; mi < 4; ++mi)
        acc[mi][ni] = wmma_bf16(af[mi], bfq[ni], acc[mi][ni]);
  }

  // C layout: lane holds (row = r + 8*lh, col = lr) within each 16x16 tile
  if (mode == 0) {
    __bf16* O = (__bf16*)Out;
#pragma unroll
    for (int mi = 0; mi < 4; ++mi)
#pragma unroll
      for (int ni = 0; ni < 4; ++ni)
#pragma unroll
        for (int r = 0; r < 8; ++r)
          O[(size_t)(mtile + mi * 16 + r + 8 * lh) * EMB + ntile + ni * 16 + lr] =
              f2bf(acc[mi][ni][r] * oscale);
  } else if (mode == 1) {
    // Vt[b][h][d][s]; rows r=0..7 are 8 consecutive s -> one 16B store
    __bf16* O = (__bf16*)Out;
#pragma unroll
    for (int mi = 0; mi < 4; ++mi) {
      const int m0 = mtile + mi * 16 + 8 * lh;
      const int bb = m0 >> 11;
      const int s  = m0 & (SEQ - 1);
#pragma unroll
      for (int ni = 0; ni < 4; ++ni) {
        const int d  = ntile + ni * 16 + lr;
        const int hh = d >> 6;
        const int dd = d & (DHEAD - 1);
        u16x8 pk;
#pragma unroll
        for (int r = 0; r < 8; ++r) pk[r] = f2bf_bits(acc[mi][ni][r] * oscale);
        *(u16x8*)(O + ((size_t)((bb * NHEAD + hh) * DHEAD + dd)) * SEQ + s) = pk;
      }
    }
  } else {
    float* O = (float*)Out;
#pragma unroll
    for (int mi = 0; mi < 4; ++mi)
#pragma unroll
      for (int ni = 0; ni < 4; ++ni)
#pragma unroll
        for (int r = 0; r < 8; ++r)
          O[(size_t)(mtile + mi * 16 + r + 8 * lh) * EMB + ntile + ni * 16 + lr] =
              acc[mi][ni][r] * oscale;
  }
}

// ---------------------------------------------------------------------------
// Flash attention (causal). One wave handles 16 query rows of one (b,h),
// iterating 64-key blocks. Q is pre-scaled by (1/sqrt(Dh))*log2(e), so the
// WMMA scores are already in the exp2 domain.
// Q, K stored [M,1024] bf16 (row = b*SEQ+s, col = h*64+d).
// V stored transposed Vt[b][h][d][s] bf16.  Output Y [M,1024] bf16.
// ---------------------------------------------------------------------------
__global__ __launch_bounds__(128)
void attn_fwd(const __bf16* __restrict__ Q, const __bf16* __restrict__ Kt,
              const __bf16* __restrict__ Vt, __bf16* __restrict__ Y) {
  __shared__ alignas(64) unsigned short ptile[4][16][64];

  const int lane  = threadIdx.x & 31;
  const int wave  = threadIdx.x >> 5;
  const int lr    = lane & 15;
  const int lh    = lane >> 4;
  const int qtile = blockIdx.x * 4 + wave;
  const int h     = blockIdx.y;
  const int b     = blockIdx.z;
  const int qbase = qtile * 16;

  // Q A-fragments for d=0..31 and d=32..63 (contraction over Dh=64)
  const __bf16* qrow = Q + ((size_t)(b * SEQ + qbase + lr)) * EMB + h * DHEAD;
  bf16x16 aq0 = cat8(*(const bf16x8*)(qrow + lh * 16),
                     *(const bf16x8*)(qrow + lh * 16 + 8));
  bf16x16 aq1 = cat8(*(const bf16x8*)(qrow + 32 + lh * 16),
                     *(const bf16x8*)(qrow + 32 + lh * 16 + 8));

  f32x8 o[4];
  float mrow[8], lrow[8];
#pragma unroll
  for (int ni = 0; ni < 4; ++ni)
#pragma unroll
    for (int r = 0; r < 8; ++r) o[ni][r] = 0.f;
#pragma unroll
  for (int r = 0; r < 8; ++r) { mrow[r] = -1e30f; lrow[r] = 0.f; }

  const int kend = ((qbase + 79) >> 6) << 6;   // keys [0,kend) in 64-blocks

  for (int kb = 0; kb < kend; kb += 64) {
    // scores 16x64 = four 16x16 C tiles; each = 2 WMMAs over Dh
    f32x8 sc4[4];
#pragma unroll
    for (int sub = 0; sub < 4; ++sub) {
      const __bf16* k0 = Kt +
          ((size_t)(b * SEQ + kb + sub * 16 + lr)) * EMB + h * DHEAD + lh * 16;
      bf16x16 bk0 = *(const bf16x16*)(k0);
      bf16x16 bk1 = *(const bf16x16*)(k0 + 32);
      f32x8 s;
#pragma unroll
      for (int r = 0; r < 8; ++r) s[r] = 0.f;
      s = wmma_bf16(aq0, bk0, s);
      s = wmma_bf16(aq1, bk1, s);
      sc4[sub] = s;
    }

    // online softmax per row (row = r + 8*lh; cols = lr + 16*sub)
    float corr[8];
#pragma unroll
    for (int r = 0; r < 8; ++r) {
      const int qg = qbase + r + 8 * lh;
      float v0 = (kb + lr      <= qg) ? sc4[0][r] : -1e30f;
      float v1 = (kb + 16 + lr <= qg) ? sc4[1][r] : -1e30f;
      float v2 = (kb + 32 + lr <= qg) ? sc4[2][r] : -1e30f;
      float v3 = (kb + 48 + lr <= qg) ? sc4[3][r] : -1e30f;
      float t = fmaxf(fmaxf(v0, v1), fmaxf(v2, v3));
      t = fmaxf(t, __shfl_xor(t, 1, 16));
      t = fmaxf(t, __shfl_xor(t, 2, 16));
      t = fmaxf(t, __shfl_xor(t, 4, 16));
      t = fmaxf(t, __shfl_xor(t, 8, 16));
      float mnew = fmaxf(mrow[r], t);
      float c = __builtin_amdgcn_exp2f(mrow[r] - mnew);
      corr[r] = c;
      mrow[r] = mnew;
      float p0 = __builtin_amdgcn_exp2f(v0 - mnew);
      float p1 = __builtin_amdgcn_exp2f(v1 - mnew);
      float p2 = __builtin_amdgcn_exp2f(v2 - mnew);
      float p3 = __builtin_amdgcn_exp2f(v3 - mnew);
      float rs = (p0 + p1) + (p2 + p3);
      rs += __shfl_xor(rs, 1, 16);
      rs += __shfl_xor(rs, 2, 16);
      rs += __shfl_xor(rs, 4, 16);
      rs += __shfl_xor(rs, 8, 16);
      lrow[r] = lrow[r] * c + rs;
      // C-layout -> LDS [row][col] as bf16 for A-fragment reload
      ptile[wave][r + 8 * lh][lr]      = f2bf_bits(p0);
      ptile[wave][r + 8 * lh][lr + 16] = f2bf_bits(p1);
      ptile[wave][r + 8 * lh][lr + 32] = f2bf_bits(p2);
      ptile[wave][r + 8 * lh][lr + 48] = f2bf_bits(p3);
    }
#pragma unroll
    for (int ni = 0; ni < 4; ++ni)
#pragma unroll
      for (int r = 0; r < 8; ++r) o[ni][r] *= corr[r];

    // P as two A-fragments 16x32 (same-wave DS ops are in-order; no barrier)
    bf16x16 ap0 = cat8(
        __builtin_bit_cast(bf16x8, *(const u16x8*)&ptile[wave][lr][lh * 16]),
        __builtin_bit_cast(bf16x8, *(const u16x8*)&ptile[wave][lr][lh * 16 + 8]));
    bf16x16 ap1 = cat8(
        __builtin_bit_cast(bf16x8, *(const u16x8*)&ptile[wave][lr][32 + lh * 16]),
        __builtin_bit_cast(bf16x8, *(const u16x8*)&ptile[wave][lr][32 + lh * 16 + 8]));

    // o[16x64] += P[16x64] * V[64x64]; Vt rows give contiguous 32B B-chunks
#pragma unroll
    for (int ni = 0; ni < 4; ++ni) {
      const __bf16* vp = Vt +
          ((size_t)((b * NHEAD + h) * DHEAD + ni * 16 + lr)) * SEQ + kb + lh * 16;
      bf16x16 bv0 = *(const bf16x16*)vp;
      bf16x16 bv1 = *(const bf16x16*)(vp + 32);
      o[ni] = wmma_bf16(ap0, bv0, o[ni]);
      o[ni] = wmma_bf16(ap1, bv1, o[ni]);
    }
  }

  // normalize and store Y [M,1024] bf16
#pragma unroll
  for (int r = 0; r < 8; ++r) {
    const float inv = __builtin_amdgcn_rcpf(lrow[r]);
    const size_t row = (size_t)(b * SEQ + qbase + r + 8 * lh) * EMB + h * DHEAD;
#pragma unroll
    for (int ni = 0; ni < 4; ++ni)
      Y[row + ni * 16 + lr] = f2bf(o[ni][r] * inv);
  }
}

// ---------------------------------------------------------------------------
// Host launcher
// ---------------------------------------------------------------------------
extern "C" void kernel_launch(void* const* d_in, const int* in_sizes, int n_in,
                              void* d_out, int out_size, void* d_ws,
                              size_t ws_size, hipStream_t stream) {
  const float* x  = (const float*)d_in[0];
  const float* Wq = (const float*)d_in[1];
  const float* Wk = (const float*)d_in[2];
  const float* Wv = (const float*)d_in[3];
  const float* Wo = (const float*)d_in[4];

  char*  ws  = (char*)d_ws;
  size_t off = 0;
  auto alloc = [&](size_t bytes) -> void* {
    void* p = ws + off;
    off += (bytes + 255) & ~(size_t)255;
    return p;
  };

  __bf16* xb  = (__bf16*)alloc((size_t)MROWS * EMB * 2);
  __bf16* wqb = (__bf16*)alloc((size_t)EMB * EMB * 2);
  __bf16* wkb = (__bf16*)alloc((size_t)EMB * EMB * 2);
  __bf16* wvb = (__bf16*)alloc((size_t)EMB * EMB * 2);
  __bf16* wob = (__bf16*)alloc((size_t)EMB * EMB * 2);
  __bf16* Qb  = (__bf16*)alloc((size_t)MROWS * EMB * 2);
  __bf16* Kb  = (__bf16*)alloc((size_t)MROWS * EMB * 2);
  __bf16* Vtb = (__bf16*)alloc((size_t)MROWS * EMB * 2);
  __bf16* Yb  = (__bf16*)alloc((size_t)MROWS * EMB * 2);

  auto cvt = [&](const float* src, __bf16* dst, int n) {
    int n8 = n / 8;
    cvt_f32_bf16_v8<<<(n8 + 255) / 256, 256, 0, stream>>>(
        src, (unsigned short*)dst, n8);
  };
  cvt(x, xb, MROWS * EMB);
  cvt(Wq, wqb, EMB * EMB);
  cvt(Wk, wkb, EMB * EMB);
  cvt(Wv, wvb, EMB * EMB);
  cvt(Wo, wob, EMB * EMB);

  // Fold softmax scale * log2(e) into Q so attention scores are exp2-domain.
  const float qscale = 0.125f * 1.4426950408889634f;

  dim3 blk(256);
  dim3 gg(EMB / 256, MROWS / 128);
  gemm_bf16<<<gg, blk, 0, stream>>>(xb, wqb, Qb, 0, qscale);  // Q (pre-scaled)
  gemm_bf16<<<gg, blk, 0, stream>>>(xb, wkb, Kb, 0, 1.0f);    // K row-major
  gemm_bf16<<<gg, blk, 0, stream>>>(xb, wvb, Vtb, 1, 1.0f);   // V transposed

  attn_fwd<<<dim3(SEQ / 16 / 4, NHEAD, BATCH), dim3(128), 0, stream>>>(
      Qb, Kb, Vtb, Yb);

  gemm_bf16<<<gg, blk, 0, stream>>>(Yb, wob, d_out, 2, 1.0f); // fp32 out
}
